// TransformerBlock_15195594293594
// MI455X (gfx1250) — matrix operand order repaired and checked by reading, not compiled, not run on hardware
//
#include <hip/hip_runtime.h>
#include <hip/hip_bf16.h>

typedef __attribute__((ext_vector_type(16))) _Float16 v16h;
typedef __attribute__((ext_vector_type(8)))  _Float16 v8h;
typedef __attribute__((ext_vector_type(8)))  float    v8f;

#define EDIM 128
#define TSEQ 64
#define NHEAD 4
#define HDIM 32
#define NBATCH 4096

// workspace layout in _Float16 units (packed B-fragment blobs, 512 halfs per 32x16 tile)
#define O_QKV 0
#define O_WO  49152     // 96 blobs * 512
#define O_W1  65536     // +32 blobs
#define O_W2  131072    // +128 blobs
#define WS_HALF_TOTAL 196608  // +128 blobs  (384 KB)

union AFrag { v16h v; v8h h[2]; };

__device__ __forceinline__ v8f wmma_f16(v16h a, v16h b, v8f c) {
    return __builtin_amdgcn_wmma_f32_16x16x32_f16(false, a, false, b, (short)0, c, false, false);
}

// A fragment (16x32 f16) from LDS, row-major with given stride (in halfs).
// lane: m = mt*16 + (lane&15); k contiguous at kbase+g*16 .. +15 (g = lane>>4)
__device__ __forceinline__ v16h load_a_lds(const _Float16* base, int stride, int m, int kbase, int g) {
    AFrag f;
    f.h[0] = *(const v8h*)(base + m * stride + kbase + g * 16);
    f.h[1] = *(const v8h*)(base + m * stride + kbase + g * 16 + 8);
    return f.v;
}

// B fragment (32x16 f16) from packed blob: contiguous 16 halfs per lane
__device__ __forceinline__ v16h load_b_pk(const _Float16* blobs, int blobIdx, int lane) {
    return *(const v16h*)(blobs + blobIdx * 512 + lane * 16);
}

// ---------------- weight packing (f32 -> f16 fragment blobs) ----------------
__global__ __launch_bounds__(256) void pack_weights(
    const float* __restrict__ Wq, const float* __restrict__ Wk, const float* __restrict__ Wv,
    const float* __restrict__ Wo, const float* __restrict__ W1, const float* __restrict__ W2,
    _Float16* __restrict__ ws) {
    int idx = blockIdx.x * 256 + threadIdx.x;
    if (idx >= WS_HALF_TOTAL) return;
    float v;
    if (idx < O_WO) {                       // WQKV fused: K=128, N=384 (Q|K|V, col = h*32+d)
        int rel = idx, blob = rel >> 9, within = rel & 511;
        int lane = within >> 4, j = within & 15;
        int kt = blob & 3, nt = blob >> 2;
        int k = kt * 32 + (lane >> 4) * 16 + j;
        int n = nt * 16 + (lane & 15);
        const float* W = (n < 128) ? Wq : (n < 256 ? Wk : Wv);
        int c = n & 127, h = c >> 5, d = c & 31;
        v = W[(h * EDIM + k) * HDIM + d];
    } else if (idx < O_W1) {                // Wo: 128x128
        int rel = idx - O_WO, blob = rel >> 9, within = rel & 511;
        int lane = within >> 4, j = within & 15;
        int kt = blob & 3, nt = blob >> 2;
        int k = kt * 32 + (lane >> 4) * 16 + j;
        int n = nt * 16 + (lane & 15);
        v = Wo[k * EDIM + n];
    } else if (idx < O_W2) {                // W1: 128x512
        int rel = idx - O_W1, blob = rel >> 9, within = rel & 511;
        int lane = within >> 4, j = within & 15;
        int kt = blob & 3, nt = blob >> 2;
        int k = kt * 32 + (lane >> 4) * 16 + j;
        int n = nt * 16 + (lane & 15);
        v = W1[k * (4 * EDIM) + n];
    } else {                                // W2: 512x128 (Kt=16)
        int rel = idx - O_W2, blob = rel >> 9, within = rel & 511;
        int lane = within >> 4, j = within & 15;
        int kt = blob & 15, nt = blob >> 4;
        int k = kt * 32 + (lane >> 4) * 16 + j;
        int n = nt * 16 + (lane & 15);
        v = W2[k * EDIM + n];
    }
    ws[idx] = (_Float16)v;
}

// ---------------- kernel 1: LN1 + causal MHA + out-proj + residual ----------------
__global__ __launch_bounds__(256) void attn_kernel(
    const float* __restrict__ x, const _Float16* __restrict__ ws,
    const float* __restrict__ bo, const float* __restrict__ g1, const float* __restrict__ be1,
    float* __restrict__ x2out) {
    __shared__ _Float16 sH[TSEQ * EDIM];     // 16KB: h, later attn-concat
    __shared__ _Float16 sQK[TSEQ * 256];     // 32KB: Q|K, later P[4][64][64]
    __shared__ _Float16 sVt[EDIM * TSEQ];    // 16KB: V^T  [d][s]

    const int tid = threadIdx.x;
    const int lane = tid & 31;
    const int wave = __builtin_amdgcn_readfirstlane(tid >> 5);  // wave-uniform -> SGPR
    const int g = lane >> 4, l15 = lane & 15;
    const int b = blockIdx.x;
    const float* xb = x + (size_t)b * (TSEQ * EDIM);

    // --- LN1: global x -> sH (f16) ---
    {
        int row = tid >> 2, part = tid & 3;
        const float* xr = xb + row * EDIM + part * 32;
        float v[32]; float s0 = 0.f, s1 = 0.f;
        #pragma unroll
        for (int i = 0; i < 32; i++) { v[i] = xr[i]; s0 += v[i]; s1 += v[i] * v[i]; }
        s0 += __shfl_xor(s0, 1, 32); s1 += __shfl_xor(s1, 1, 32);
        s0 += __shfl_xor(s0, 2, 32); s1 += __shfl_xor(s1, 2, 32);
        float mu = s0 * (1.0f / EDIM);
        float var = s1 * (1.0f / EDIM) - mu * mu;
        float rs = rsqrtf(var + 1e-5f);
        #pragma unroll
        for (int i = 0; i < 32; i++) {
            int c = part * 32 + i;
            sH[row * EDIM + c] = (_Float16)((v[i] - mu) * rs * g1[c] + be1[c]);
        }
    }
    __syncthreads();

    // --- QKV GEMM: [64x128] x [128x384], 96 C-tiles over 8 waves ---
    for (int i = 0; i < 12; i++) {
        const int idx = wave * 12 + i;          // scalar
        const int mt = idx / 24, nt = idx % 24; // scalar div/mod
        v8f acc = {};
        #pragma unroll
        for (int kt = 0; kt < 4; kt++) {
            if (kt < 3) __builtin_prefetch(ws + O_QKV + (nt * 4 + kt + 1) * 512 + lane * 16, 0, 0);
            v16h a = load_a_lds(sH, EDIM, mt * 16 + l15, kt * 32, g);
            v16h bb = load_b_pk(ws + O_QKV, nt * 4 + kt, lane);
            acc = wmma_f16(a, bb, acc);
        }
        if (nt < 16) {                          // scalar branch: Q|K tiles
            int ng = nt * 16 + l15;
            #pragma unroll
            for (int r = 0; r < 8; r++) {
                int m = mt * 16 + g * 8 + r;
                sQK[m * 256 + ng] = (_Float16)acc[r];
            }
        } else {                                // V tiles, stored transposed
            int d = (nt - 16) * 16 + l15;
            #pragma unroll
            for (int r = 0; r < 8; r++) {
                int m = mt * 16 + g * 8 + r;
                sVt[d * TSEQ + m] = (_Float16)acc[r];
            }
        }
    }
    __syncthreads();

    // --- scores + causal softmax (in regs) + P -> LDS + attn = P*V ---
    {
        const int hh = wave >> 1;       // head (scalar)
        const int mhalf = wave & 1;     // which pair of m-tiles (scalar)
        float P[2][4][8];
        #pragma unroll
        for (int mi = 0; mi < 2; mi++) {
            const int mt = mhalf * 2 + mi;
            v8f acc[4];
            #pragma unroll
            for (int nt = 0; nt < 4; nt++) {
                v16h a = load_a_lds(sQK, 256, mt * 16 + l15, hh * 32, g);
                // B = K^T: lane col n -> row s of K; k=d contiguous at col 128+hh*32+g*16
                v16h bb = *(const v16h*)(sQK + (nt * 16 + l15) * 256 + 128 + hh * 32 + g * 16);
                v8f z = {};
                acc[nt] = wmma_f16(a, bb, z);
            }
            const float scale = 0.17677669529663687f;  // 1/sqrt(32)
            #pragma unroll
            for (int r = 0; r < 8; r++) {
                int t = mt * 16 + g * 8 + r;
                float vals[4]; float mx = -1e30f;
                #pragma unroll
                for (int nt = 0; nt < 4; nt++) {
                    int s = nt * 16 + l15;
                    float sc = acc[nt][r] * scale;
                    vals[nt] = (s <= t) ? sc : -1e30f;
                    mx = fmaxf(mx, vals[nt]);
                }
                #pragma unroll
                for (int off = 1; off < 16; off <<= 1) mx = fmaxf(mx, __shfl_xor(mx, off, 32));
                float sum = 0.f;
                #pragma unroll
                for (int nt = 0; nt < 4; nt++) { vals[nt] = __expf(vals[nt] - mx); sum += vals[nt]; }
                #pragma unroll
                for (int off = 1; off < 16; off <<= 1) sum += __shfl_xor(sum, off, 32);
                float inv = 1.0f / sum;
                #pragma unroll
                for (int nt = 0; nt < 4; nt++) P[mi][nt][r] = vals[nt] * inv;
            }
        }
        __syncthreads();  // all waves done reading Q,K
        _Float16* sP = sQK + hh * (64 * 64);  // reuse Q|K region as P[4][64][64]
        #pragma unroll
        for (int mi = 0; mi < 2; mi++) {
            const int mt = mhalf * 2 + mi;
            #pragma unroll
            for (int nt = 0; nt < 4; nt++)
                #pragma unroll
                for (int r = 0; r < 8; r++)
                    sP[(mt * 16 + g * 8 + r) * 64 + nt * 16 + l15] = (_Float16)P[mi][nt][r];
        }
        __syncthreads();
        // attn = P[64x64] * V[64x32] -> sH cols hh*32..+31
        #pragma unroll
        for (int mi = 0; mi < 2; mi++) {
            const int mt = mhalf * 2 + mi;
            #pragma unroll
            for (int nt = 0; nt < 2; nt++) {
                v8f acc = {};
                #pragma unroll
                for (int kt = 0; kt < 2; kt++) {
                    v16h a = load_a_lds(sP, 64, mt * 16 + l15, kt * 32, g);
                    int d = hh * 32 + nt * 16 + l15;
                    v16h bb = *(const v16h*)(sVt + d * TSEQ + kt * 32 + g * 16);
                    acc = wmma_f16(a, bb, acc);
                }
                #pragma unroll
                for (int r = 0; r < 8; r++) {
                    int m = mt * 16 + g * 8 + r;
                    sH[m * EDIM + hh * 32 + nt * 16 + l15] = (_Float16)acc[r];
                }
            }
        }
    }
    __syncthreads();

    // --- out-proj + bias + residual -> x2 (global) ---
    #pragma unroll
    for (int i = 0; i < 4; i++) {
        const int idx = wave * 4 + i;
        const int mt = idx >> 3, nt = idx & 7;  // scalar
        v8f acc = {};
        #pragma unroll
        for (int kt = 0; kt < 4; kt++) {
            if (kt < 3) __builtin_prefetch(ws + O_WO + (nt * 4 + kt + 1) * 512 + lane * 16, 0, 0);
            v16h a = load_a_lds(sH, EDIM, mt * 16 + l15, kt * 32, g);
            v16h bb = load_b_pk(ws + O_WO, nt * 4 + kt, lane);
            acc = wmma_f16(a, bb, acc);
        }
        int n = nt * 16 + l15;
        float bias = bo[n];
        #pragma unroll
        for (int r = 0; r < 8; r++) {
            int m = mt * 16 + g * 8 + r;
            x2out[(size_t)b * (TSEQ * EDIM) + m * EDIM + n] = acc[r] + bias + xb[m * EDIM + n];
        }
    }
}

// ---------------- kernel 2: LN2 + FFN (GELU exact) + residual, in-place on d_out ----------------
__global__ __launch_bounds__(256) void ffn_kernel(
    const _Float16* __restrict__ ws,
    const float* __restrict__ b1, const float* __restrict__ b2,
    const float* __restrict__ g2, const float* __restrict__ be2,
    float* __restrict__ xio) {
    __shared__ float    sX2[TSEQ * EDIM];   // 32KB
    __shared__ _Float16 sH2[TSEQ * EDIM];   // 16KB
    __shared__ _Float16 sFC[TSEQ * EDIM];   // 16KB (hidden chunk)

    const int tid = threadIdx.x;
    const int lane = tid & 31;
    const int wave = __builtin_amdgcn_readfirstlane(tid >> 5);  // scalar
    const int g = lane >> 4, l15 = lane & 15;
    const int b = blockIdx.x;
    float* xb = xio + (size_t)b * (TSEQ * EDIM);

    // load x2 tile
    {
        const float4* src = (const float4*)xb;
        float4* dst = (float4*)sX2;
        for (int i = tid; i < TSEQ * EDIM / 4; i += 256) dst[i] = src[i];
    }
    __syncthreads();

    // LN2
    {
        int row = tid >> 2, part = tid & 3;
        float v[32]; float s0 = 0.f, s1 = 0.f;
        #pragma unroll
        for (int i = 0; i < 32; i++) { v[i] = sX2[row * EDIM + part * 32 + i]; s0 += v[i]; s1 += v[i] * v[i]; }
        s0 += __shfl_xor(s0, 1, 32); s1 += __shfl_xor(s1, 1, 32);
        s0 += __shfl_xor(s0, 2, 32); s1 += __shfl_xor(s1, 2, 32);
        float mu = s0 * (1.0f / EDIM);
        float var = s1 * (1.0f / EDIM) - mu * mu;
        float rs = rsqrtf(var + 1e-5f);
        #pragma unroll
        for (int i = 0; i < 32; i++) {
            int c = part * 32 + i;
            sH2[row * EDIM + c] = (_Float16)((v[i] - mu) * rs * g2[c] + be2[c]);
        }
    }
    __syncthreads();

    // FFN: hidden 512 in 4 chunks of 128; output accumulators persist in regs
    v8f oacc[4];
    #pragma unroll
    for (int i = 0; i < 4; i++) oacc[i] = (v8f){};

    for (int c = 0; c < 4; c++) {
        // FFN1 chunk: gelu(h2 @ W1[:, c*128 : c*128+128] + b1) -> sFC
        #pragma unroll
        for (int i = 0; i < 4; i++) {
            const int idx = wave * 4 + i;
            const int mt = idx >> 3, nt = idx & 7;  // scalar
            v8f acc = {};
            #pragma unroll
            for (int kt = 0; kt < 4; kt++) {
                if (kt < 3) __builtin_prefetch(ws + O_W1 + ((c * 8 + nt) * 4 + kt + 1) * 512 + lane * 16, 0, 0);
                v16h a = load_a_lds(sH2, EDIM, mt * 16 + l15, kt * 32, g);
                v16h bb = load_b_pk(ws + O_W1, (c * 8 + nt) * 4 + kt, lane);
                acc = wmma_f16(a, bb, acc);
            }
            int nloc = nt * 16 + l15;
            float bias = b1[c * EDIM + nloc];
            #pragma unroll
            for (int r = 0; r < 8; r++) {
                float vv = acc[r] + bias;
                float ge = 0.5f * vv * (1.0f + erff(vv * 0.70710678118654752f));
                sFC[(mt * 16 + g * 8 + r) * EDIM + nloc] = (_Float16)ge;
            }
        }
        __syncthreads();
        // FFN2 partial: accumulate sFC @ W2[c*128 : c*128+128, :]
        #pragma unroll
        for (int i = 0; i < 4; i++) {
            const int idx = wave * 4 + i;
            const int mt = idx >> 3, nt = idx & 7;  // scalar
            #pragma unroll
            for (int kt = 0; kt < 4; kt++) {
                int ktg = c * 4 + kt;
                if (kt < 3) __builtin_prefetch(ws + O_W2 + (nt * 16 + ktg + 1) * 512 + lane * 16, 0, 0);
                v16h a = load_a_lds(sFC, EDIM, mt * 16 + l15, kt * 32, g);
                v16h bb = load_b_pk(ws + O_W2, nt * 16 + ktg, lane);
                oacc[i] = wmma_f16(a, bb, oacc[i]);
            }
        }
        __syncthreads();  // WAR before next chunk overwrites sFC
    }

    // finalize: + b2 + residual, overwrite d_out
    #pragma unroll
    for (int i = 0; i < 4; i++) {
        const int idx = wave * 4 + i;
        const int mt = idx >> 3, nt = idx & 7;
        int n = nt * 16 + l15;
        float bias = b2[n];
        #pragma unroll
        for (int r = 0; r < 8; r++) {
            int m = mt * 16 + g * 8 + r;
            xb[m * EDIM + n] = oacc[i][r] + bias + sX2[m * EDIM + n];
        }
    }
}

extern "C" void kernel_launch(void* const* d_in, const int* in_sizes, int n_in,
                              void* d_out, int out_size, void* d_ws, size_t ws_size,
                              hipStream_t stream) {
    const float* x   = (const float*)d_in[0];
    const float* Wq  = (const float*)d_in[1];
    const float* Wk  = (const float*)d_in[2];
    const float* Wv  = (const float*)d_in[3];
    const float* Wo  = (const float*)d_in[4];
    const float* bo  = (const float*)d_in[5];
    const float* W1  = (const float*)d_in[6];
    const float* b1  = (const float*)d_in[7];
    const float* W2  = (const float*)d_in[8];
    const float* b2  = (const float*)d_in[9];
    const float* g1  = (const float*)d_in[10];
    const float* be1 = (const float*)d_in[11];
    const float* g2  = (const float*)d_in[12];
    const float* be2 = (const float*)d_in[13];
    float* out = (float*)d_out;
    _Float16* wsh = (_Float16*)d_ws;

    pack_weights<<<(WS_HALF_TOTAL + 255) / 256, 256, 0, stream>>>(Wq, Wk, Wv, Wo, W1, W2, wsh);
    attn_kernel<<<NBATCH, 256, 0, stream>>>(x, wsh, bo, g1, be1, out);
    ffn_kernel<<<NBATCH, 256, 0, stream>>>(wsh, b1, b2, g2, be2, out);
}
